// MultiHeadAttention_23029614641748
// MI455X (gfx1250) — matrix operand order repaired and checked by reading, not compiled, not run on hardware
//
#include <hip/hip_runtime.h>
#include <hip/hip_bf16.h>

#define HIDDEN 1024
#define HEADS 16
#define HEAD_DIM 64
#define BATCH 4
#define SEQ 2048
#define LN_EPS 1e-5f

typedef __attribute__((ext_vector_type(16))) __bf16 v16bf;
typedef __attribute__((ext_vector_type(16))) unsigned short v16us;
typedef __attribute__((ext_vector_type(8))) float v8f;

__device__ __forceinline__ unsigned short f32_to_bf16(float f) {
  unsigned int u = __builtin_bit_cast(unsigned int, f);
  u += 0x7FFFu + ((u >> 16) & 1u);  // round-to-nearest-even
  return (unsigned short)(u >> 16);
}

// K-index map within a 32-wide chunk for the 16-bit A fragment (ISA 7.12.2):
// lanes 0-15: VGPR0..3 -> K0..7, VGPR4..7 -> K16..23 ; lanes 16-31 get +8
__device__ __forceinline__ int amap(int e, int half) {
  return e + 8 * half + 8 * (e >> 3);
}

#define A_STRIDE 34  // ushort stride of staged A tiles ([rows][32] + pad)
#define B_STRIDE 72  // ushort stride of staged B tiles ([32][64] + pad)

// ---------------------------------------------------------------------------
// Per-wave 32x32 WMMA K-step (K chunk of 32) on staged LDS tiles:
// 2 A fragments x 2 B fragments -> 4 v_wmma_f32_16x16x32_bf16.
// ---------------------------------------------------------------------------
__device__ __forceinline__ void wave_step(const unsigned short* __restrict__ Ab,
                                          const unsigned short* __restrict__ Bb,
                                          int mrow, int wn, int l16, int half,
                                          v8f (&acc)[2][2]) {
  v16us au[2], bu[2];
#pragma unroll
  for (int e = 0; e < 16; ++e) {
    int ka = amap(e, half);
    au[0][e] = Ab[(mrow + l16) * A_STRIDE + ka];
    au[1][e] = Ab[(mrow + 16 + l16) * A_STRIDE + ka];
    int kr = amap(e, half);  // B: same K map as A (table order)
    bu[0][e] = Bb[kr * B_STRIDE + wn + l16];
    bu[1][e] = Bb[kr * B_STRIDE + wn + 16 + l16];
  }
#pragma unroll
  for (int mi = 0; mi < 2; ++mi) {
    v16bf Af = __builtin_bit_cast(v16bf, au[mi]);
#pragma unroll
    for (int ni = 0; ni < 2; ++ni) {
      v16bf Bf = __builtin_bit_cast(v16bf, bu[ni]);
      acc[mi][ni] = __builtin_amdgcn_wmma_f32_16x16x32_bf16(
          false, Af, false, Bf, (short)0, acc[mi][ni], false, false);
    }
  }
}

// ---------------------------------------------------------------------------
// Tiled GEMM: C[M,N] = A[M,K](f32) @ W[K,N](f32) + bias[N]
// Block tile 128x64, 8 waves of 32x32; double-buffered LDS with register
// staging: one __syncthreads per K-step, global loads issued ahead of WMMA.
// ---------------------------------------------------------------------------
template <bool OUT_BF16>
__global__ __launch_bounds__(256) void gemm_bias_kernel(
    const float* __restrict__ A, const float* __restrict__ W,
    const float* __restrict__ bias, void* __restrict__ Cout,
    int M, int N, int K) {
  __shared__ unsigned short Albs[2][128][A_STRIDE];
  __shared__ unsigned short Blds[2][32][B_STRIDE];
  const int tid = threadIdx.x;
  const int lane = tid & 31, wave = tid >> 5;
  const int half = lane >> 4, l16 = lane & 15;
  const int wm = wave >> 1;        // 4 row groups of 32
  const int wn = (wave & 1) * 32;  // 2 col halves of 32
  const int m0 = blockIdx.y * 128;
  const int n0 = blockIdx.x * 64;

  v8f acc[2][2] = {{{}, {}}, {{}, {}}};
  float areg[16], breg[8];

  // Preload K-chunk 0 and stage to LDS buffer 0.
#pragma unroll
  for (int i = 0; i < 16; ++i) {  // A tile 128x32
    int idx = tid + i * 256;
    int r = idx >> 5, c = idx & 31;
    areg[i] = A[(size_t)(m0 + r) * K + c];
  }
#pragma unroll
  for (int i = 0; i < 8; ++i) {  // W tile 32x64
    int idx = tid + i * 256;
    int r = idx >> 6, c = idx & 63;
    breg[i] = W[(size_t)r * N + n0 + c];
  }
#pragma unroll
  for (int i = 0; i < 16; ++i) {
    int idx = tid + i * 256;
    Albs[0][idx >> 5][idx & 31] = f32_to_bf16(areg[i]);
  }
#pragma unroll
  for (int i = 0; i < 8; ++i) {
    int idx = tid + i * 256;
    Blds[0][idx >> 6][idx & 63] = f32_to_bf16(breg[i]);
  }
  __syncthreads();

  const int nsteps = K >> 5;
  for (int s = 0; s < nsteps; ++s) {
    const int cur = s & 1;
    const bool more = (s + 1) < nsteps;
    if (more) {  // issue next chunk's global loads before computing
      const int kk = (s + 1) << 5;
#pragma unroll
      for (int i = 0; i < 16; ++i) {
        int idx = tid + i * 256;
        int r = idx >> 5, c = idx & 31;
        areg[i] = A[(size_t)(m0 + r) * K + kk + c];
      }
#pragma unroll
      for (int i = 0; i < 8; ++i) {
        int idx = tid + i * 256;
        int r = idx >> 6, c = idx & 63;
        breg[i] = W[(size_t)(kk + r) * N + n0 + c];
      }
      if (s + 2 < nsteps)
        __builtin_prefetch(&A[(size_t)(m0 + (tid >> 1)) * K + ((s + 2) << 5)], 0, 3);
    }
    wave_step(&Albs[cur][0][0], &Blds[cur][0][0], 32 * wm, wn, l16, half, acc);
    if (more) {
      const int nxt = cur ^ 1;
#pragma unroll
      for (int i = 0; i < 16; ++i) {
        int idx = tid + i * 256;
        Albs[nxt][idx >> 5][idx & 31] = f32_to_bf16(areg[i]);
      }
#pragma unroll
      for (int i = 0; i < 8; ++i) {
        int idx = tid + i * 256;
        Blds[nxt][idx >> 6][idx & 63] = f32_to_bf16(breg[i]);
      }
      __syncthreads();
    }
  }

#pragma unroll
  for (int mi = 0; mi < 2; ++mi)
#pragma unroll
    for (int ni = 0; ni < 2; ++ni)
#pragma unroll
      for (int v = 0; v < 8; ++v) {  // C layout: VGPR v -> M=v+8*half, N=lane%16
        int row = m0 + 32 * wm + mi * 16 + v + 8 * half;
        int col = n0 + wn + ni * 16 + l16;
        float r = acc[mi][ni][v] + bias[col];
        if (OUT_BF16)
          ((unsigned short*)Cout)[(size_t)row * N + col] = f32_to_bf16(r);
        else
          ((float*)Cout)[(size_t)row * N + col] = r;
      }
}

// ---------------------------------------------------------------------------
// Attention scores: per (b,h)  P = Qh @ Kh^T * (1/sqrt(hd)), masked.
// Qp/Kp are bf16 in [B,S,D] layout (head h = cols h*64..h*64+63). K=64 fits
// in one LDS stage; each wave computes a 16x32 slice of a 64x64 tile.
// ---------------------------------------------------------------------------
__global__ __launch_bounds__(256) void attn_scores_kernel(
    const unsigned short* __restrict__ Qp, const unsigned short* __restrict__ Kp,
    const int* __restrict__ mask, float* __restrict__ attn) {
  __shared__ unsigned short Qlds[64][72];
  __shared__ unsigned short Klds[64][72];
  const int tid = threadIdx.x;
  const int lane = tid & 31, wave = tid >> 5;
  const int half = lane >> 4, l16 = lane & 15;
  const int wm = wave >> 1;
  const int wn = (wave & 1) * 32;
  const int z = blockIdx.z;  // b*HEADS + h
  const int b = z / HEADS, h = z % HEADS;
  const int m0 = blockIdx.y * 64;  // query block
  const int n0 = blockIdx.x * 64;  // key block

#pragma unroll
  for (int i = 0; i < 16; ++i) {  // 64x64 bf16 tiles of Q and K
    int idx = tid + i * 256;
    int r = idx >> 6, c = idx & 63;
    Qlds[r][c] = Qp[((size_t)b * SEQ + m0 + r) * HIDDEN + h * HEAD_DIM + c];
    Klds[r][c] = Kp[((size_t)b * SEQ + n0 + r) * HIDDEN + h * HEAD_DIM + c];
  }
  __syncthreads();

  v8f acc0 = {}, acc1 = {};
#pragma unroll
  for (int kk = 0; kk < HEAD_DIM; kk += 32) {
    v16us au;
    const int m = wm * 16 + l16;
#pragma unroll
    for (int e = 0; e < 16; ++e) au[e] = Qlds[m][kk + amap(e, half)];
    v16bf a = __builtin_bit_cast(v16bf, au);

    v16us bu0, bu1;
#pragma unroll
    for (int e = 0; e < 16; ++e) {  // B[k][n] = Kh[n][k]: transpose via indexing
      int kr = kk + amap(e, half);
      bu0[e] = Klds[wn + l16][kr];
      bu1[e] = Klds[wn + 16 + l16][kr];
    }
    v16bf b0 = __builtin_bit_cast(v16bf, bu0);
    v16bf b1 = __builtin_bit_cast(v16bf, bu1);

    acc0 = __builtin_amdgcn_wmma_f32_16x16x32_bf16(false, a, false, b0,
                                                   (short)0, acc0, false, false);
    acc1 = __builtin_amdgcn_wmma_f32_16x16x32_bf16(false, a, false, b1,
                                                   (short)0, acc1, false, false);
  }

  const float scale = 0.125f;  // 1/sqrt(64)
#pragma unroll
  for (int v = 0; v < 8; ++v) {
    int qrow = m0 + wm * 16 + v + 8 * half;
    int kc0 = n0 + wn + l16, kc1 = kc0 + 16;
    size_t mbase = ((size_t)b * SEQ + qrow) * SEQ;
    size_t abase = ((size_t)z * SEQ + qrow) * SEQ;
    attn[abase + kc0] = mask[mbase + kc0] ? acc0[v] * scale : -1e9f;
    attn[abase + kc1] = mask[mbase + kc1] ? acc1[v] * scale : -1e9f;
  }
}

// ---------------------------------------------------------------------------
// Row softmax over S=2048, one block per row (bandwidth-bound).
// ---------------------------------------------------------------------------
__global__ __launch_bounds__(256) void softmax_kernel(float* __restrict__ attn) {
  __shared__ float red[256];
  const int tid = threadIdx.x;
  float* row = attn + (size_t)blockIdx.x * SEQ;
  float vals[8];
  float mx = -3.4e38f;
#pragma unroll
  for (int i = 0; i < 8; ++i) {
    vals[i] = row[tid + i * 256];
    mx = fmaxf(mx, vals[i]);
  }
  red[tid] = mx;
  __syncthreads();
  for (int s = 128; s > 0; s >>= 1) {
    if (tid < s) red[tid] = fmaxf(red[tid], red[tid + s]);
    __syncthreads();
  }
  mx = red[0];
  __syncthreads();
  float sum = 0.f;
#pragma unroll
  for (int i = 0; i < 8; ++i) {
    vals[i] = __expf(vals[i] - mx);
    sum += vals[i];
  }
  red[tid] = sum;
  __syncthreads();
  for (int s = 128; s > 0; s >>= 1) {
    if (tid < s) red[tid] += red[tid + s];
    __syncthreads();
  }
  float inv = 1.0f / red[0];
#pragma unroll
  for (int i = 0; i < 8; ++i) row[tid + i * 256] = vals[i] * inv;
}

// ---------------------------------------------------------------------------
// Context: per (b,h)  ctx = attn(f32, [S,S]) @ Vh(bf16, [S,64]).
// Same 128x64 double-buffered WMMA core; A converted f32->bf16 on staging.
// ---------------------------------------------------------------------------
__global__ __launch_bounds__(256) void attn_context_kernel(
    const float* __restrict__ attn, const unsigned short* __restrict__ Vp,
    float* __restrict__ ctx) {
  __shared__ unsigned short Albs[2][128][A_STRIDE];
  __shared__ unsigned short Blds[2][32][B_STRIDE];
  const int tid = threadIdx.x;
  const int lane = tid & 31, wave = tid >> 5;
  const int half = lane >> 4, l16 = lane & 15;
  const int wm = wave >> 1;
  const int wn = (wave & 1) * 32;
  const int z = blockIdx.z;
  const int b = z / HEADS, h = z % HEADS;
  const int m0 = blockIdx.y * 128;

  v8f acc[2][2] = {{{}, {}}, {{}, {}}};
  float areg[16];
  unsigned short breg[8];

#pragma unroll
  for (int i = 0; i < 16; ++i) {  // attn tile 128x32
    int idx = tid + i * 256;
    int r = idx >> 5, c = idx & 31;
    areg[i] = attn[((size_t)z * SEQ + m0 + r) * SEQ + c];
  }
#pragma unroll
  for (int i = 0; i < 8; ++i) {  // V tile 32x64 (already bf16)
    int idx = tid + i * 256;
    int r = idx >> 6, c = idx & 63;
    breg[i] = Vp[((size_t)b * SEQ + r) * HIDDEN + h * HEAD_DIM + c];
  }
#pragma unroll
  for (int i = 0; i < 16; ++i) {
    int idx = tid + i * 256;
    Albs[0][idx >> 5][idx & 31] = f32_to_bf16(areg[i]);
  }
#pragma unroll
  for (int i = 0; i < 8; ++i) {
    int idx = tid + i * 256;
    Blds[0][idx >> 6][idx & 63] = breg[i];
  }
  __syncthreads();

  const int nsteps = SEQ >> 5;  // 64
  for (int s = 0; s < nsteps; ++s) {
    const int cur = s & 1;
    const bool more = (s + 1) < nsteps;
    if (more) {
      const int kk = (s + 1) << 5;
#pragma unroll
      for (int i = 0; i < 16; ++i) {
        int idx = tid + i * 256;
        int r = idx >> 5, c = idx & 31;
        areg[i] = attn[((size_t)z * SEQ + m0 + r) * SEQ + kk + c];
      }
#pragma unroll
      for (int i = 0; i < 8; ++i) {
        int idx = tid + i * 256;
        int r = idx >> 6, c = idx & 63;
        breg[i] = Vp[((size_t)b * SEQ + kk + r) * HIDDEN + h * HEAD_DIM + c];
      }
      if (s + 2 < nsteps)
        __builtin_prefetch(
            &attn[((size_t)z * SEQ + m0 + (tid >> 1)) * SEQ + ((s + 2) << 5)], 0, 3);
    }
    wave_step(&Albs[cur][0][0], &Blds[cur][0][0], 32 * wm, wn, l16, half, acc);
    if (more) {
      const int nxt = cur ^ 1;
#pragma unroll
      for (int i = 0; i < 16; ++i) {
        int idx = tid + i * 256;
        Albs[nxt][idx >> 5][idx & 31] = f32_to_bf16(areg[i]);
      }
#pragma unroll
      for (int i = 0; i < 8; ++i) {
        int idx = tid + i * 256;
        Blds[nxt][idx >> 6][idx & 63] = breg[i];
      }
      __syncthreads();
    }
  }

#pragma unroll
  for (int mi = 0; mi < 2; ++mi)
#pragma unroll
    for (int ni = 0; ni < 2; ++ni)
#pragma unroll
      for (int v = 0; v < 8; ++v) {
        int row = m0 + 32 * wm + mi * 16 + v + 8 * half;
        int col = wn + ni * 16 + l16;
        ctx[((size_t)b * SEQ + row) * HIDDEN + h * HEAD_DIM + col] = acc[mi][ni][v];
      }
}

// ---------------------------------------------------------------------------
// Fused residual add + LayerNorm over D=1024, one block per row.
// ---------------------------------------------------------------------------
__global__ __launch_bounds__(256) void add_ln_kernel(
    const float* __restrict__ proj, const float* __restrict__ resid,
    const float* __restrict__ gamma, const float* __restrict__ beta,
    float* __restrict__ out) {
  __shared__ float red[256];
  const int tid = threadIdx.x;
  const size_t base = (size_t)blockIdx.x * HIDDEN;
  float x[4];
  float s = 0.f;
#pragma unroll
  for (int i = 0; i < 4; ++i) {
    int c = tid + i * 256;
    x[i] = proj[base + c] + resid[base + c];
    s += x[i];
  }
  red[tid] = s;
  __syncthreads();
  for (int st = 128; st > 0; st >>= 1) {
    if (tid < st) red[tid] += red[tid + st];
    __syncthreads();
  }
  float mu = red[0] * (1.0f / HIDDEN);
  __syncthreads();
  float v = 0.f;
#pragma unroll
  for (int i = 0; i < 4; ++i) {
    float d = x[i] - mu;
    v += d * d;
  }
  red[tid] = v;
  __syncthreads();
  for (int st = 128; st > 0; st >>= 1) {
    if (tid < st) red[tid] += red[tid + st];
    __syncthreads();
  }
  float inv = rsqrtf(red[0] * (1.0f / HIDDEN) + LN_EPS);
#pragma unroll
  for (int i = 0; i < 4; ++i) {
    int c = tid + i * 256;
    out[base + c] = (x[i] - mu) * inv * gamma[c] + beta[c];
  }
}

// ---------------------------------------------------------------------------
extern "C" void kernel_launch(void* const* d_in, const int* in_sizes, int n_in,
                              void* d_out, int out_size, void* d_ws, size_t ws_size,
                              hipStream_t stream) {
  const float* q = (const float*)d_in[0];
  const float* k = (const float*)d_in[1];
  const float* v = (const float*)d_in[2];
  const int* mask = (const int*)d_in[3];
  const float* Wq = (const float*)d_in[4];
  const float* bq = (const float*)d_in[5];
  const float* Wk = (const float*)d_in[6];
  const float* bk = (const float*)d_in[7];
  const float* Wv = (const float*)d_in[8];
  const float* bv = (const float*)d_in[9];
  const float* Wo = (const float*)d_in[10];
  const float* bo = (const float*)d_in[11];
  const float* gamma = (const float*)d_in[12];
  const float* beta = (const float*)d_in[13];

  const size_t BS = (size_t)BATCH * SEQ;  // 8192 rows
  float* out = (float*)d_out;             // [B,S,D]
  float* attn = out + BS * HIDDEN;        // [B,H,S,S]

  // Workspace (80 MB): Qp|Kp|Vp bf16 (16 MB each), ctx f32 (32 MB);
  // proj f32 (32 MB) aliases Qp+Kp which are dead by the out-projection.
  unsigned short* Qp = (unsigned short*)d_ws;
  unsigned short* Kp = Qp + BS * HIDDEN;
  unsigned short* Vp = Kp + BS * HIDDEN;
  float* ctx = (float*)(Vp + BS * HIDDEN);
  float* proj = (float*)d_ws;

  dim3 blk(256);
  dim3 gproj(HIDDEN / 64, (unsigned)(BS / 128));

  gemm_bias_kernel<true><<<gproj, blk, 0, stream>>>(q, Wq, bq, Qp, (int)BS, HIDDEN, HIDDEN);
  gemm_bias_kernel<true><<<gproj, blk, 0, stream>>>(k, Wk, bk, Kp, (int)BS, HIDDEN, HIDDEN);
  gemm_bias_kernel<true><<<gproj, blk, 0, stream>>>(v, Wv, bv, Vp, (int)BS, HIDDEN, HIDDEN);

  dim3 gsc(SEQ / 64, SEQ / 64, BATCH * HEADS);
  attn_scores_kernel<<<gsc, blk, 0, stream>>>(Qp, Kp, mask, attn);

  softmax_kernel<<<dim3((unsigned)(BATCH * HEADS * SEQ)), blk, 0, stream>>>(attn);

  dim3 gctx(1, SEQ / 128, BATCH * HEADS);
  attn_context_kernel<<<gctx, blk, 0, stream>>>(attn, Vp, ctx);

  gemm_bias_kernel<false><<<gproj, blk, 0, stream>>>(ctx, Wo, bo, proj, (int)BS, HIDDEN, HIDDEN);

  add_ln_kernel<<<dim3((unsigned)BS), blk, 0, stream>>>(proj, q, gamma, beta, out);
}
